// SimpleInteractionBlock_65171833750294
// MI455X (gfx1250) — matrix operand order repaired and checked by reading, not compiled, not run on hardware
//
#include <hip/hip_runtime.h>
#include <hip/hip_bf16.h>

typedef __attribute__((ext_vector_type(16))) _Float16 v16h;
typedef __attribute__((ext_vector_type(8)))  _Float16 v8h;
typedef __attribute__((ext_vector_type(8)))  float    v8f;

#define NRAD 8

// padded LDS row strides (halfs) -> distinct banks for the 16 rows of a frag
#define WSK_STRIDE 328   // 320 + 8
#define WL_STRIDE  72    // 64 + 8
#define LIN_STRIDE 40    // 32 + 8

// ---------------------------------------------------------------------------
// Kernel 1: zero the message accumulator planes (4 planes of N x 64 floats)
// ---------------------------------------------------------------------------
__global__ void zero_kernel(float4* __restrict__ p, long n4) {
    long i = (long)blockIdx.x * blockDim.x + threadIdx.x;
    long stride = (long)gridDim.x * blockDim.x;
    float4 z = make_float4(0.f, 0.f, 0.f, 0.f);
    for (; i < n4; i += stride) p[i] = z;
}

// ---------------------------------------------------------------------------
// Kernel 2: per-edge tensor product + scatter-add (1 wave per edge)
//   msgs layout: plane c in {0: msg0, 1..3: msg1[:,:,i]}, each [N][64] f32
// ---------------------------------------------------------------------------
__global__ __launch_bounds__(256) void edge_kernel(
    const float* __restrict__ edge_attrs,     // [E][4]
    const float* __restrict__ edge_feats,     // [E][8]
    const long long* __restrict__ edge_index, // [2][E]
    const float* __restrict__ node_feats,     // [N][128]
    const float* __restrict__ Wr,             // [8][128]
    float* __restrict__ msgs,                 // [4][N][64]
    int E, int N)
{
    int lane = threadIdx.x & 31;
    long e = (long)blockIdx.x * 8 + (threadIdx.x >> 5);
    if (e >= E) return;

    long snd = edge_index[e];
    long rcv = edge_index[(long)E + e];

    float ef[NRAD];
#pragma unroll
    for (int k = 0; k < NRAD; ++k) ef[k] = edge_feats[e * NRAD + k];

    float w1 = 0.f, w2 = 0.f, w3 = 0.f, w4 = 0.f;
#pragma unroll
    for (int k = 0; k < NRAD; ++k) {
        const float* wr = Wr + k * 128;
        w1 = fmaf(ef[k], wr[lane],       w1);
        w2 = fmaf(ef[k], wr[32 + lane],  w2);
        w3 = fmaf(ef[k], wr[64 + lane],  w3);
        w4 = fmaf(ef[k], wr[96 + lane],  w4);
    }
    const float rs8 = 0.35355339059327f;   // 1/sqrt(8)
    w1 *= rs8; w2 *= rs8; w3 *= rs8; w4 *= rs8;

    float e0  = edge_attrs[e * 4 + 0];
    float e1x = edge_attrs[e * 4 + 1];
    float e1y = edge_attrs[e * 4 + 2];
    float e1z = edge_attrs[e * 4 + 3];

    const float* nf = node_feats + (size_t)snd * 128;
    float ss = nf[lane];
    float vx = nf[32 + lane * 3 + 0];
    float vy = nf[32 + lane * 3 + 1];
    float vz = nf[32 + lane * 3 + 2];

    float dve = (vx * e1x + vy * e1y + vz * e1z) * 0.57735026919f; // /sqrt(3)

    size_t plane = (size_t)N * 64;
    float* p0 = msgs + (size_t)rcv * 64;

    float m0a = w1 * ss * e0;
    float m0b = w4 * dve;
    float w2s = w2 * ss;
    float w3e = w3 * e0;

    __hip_atomic_fetch_add(p0 + lane,       m0a, __ATOMIC_RELAXED, __HIP_MEMORY_SCOPE_AGENT);
    __hip_atomic_fetch_add(p0 + 32 + lane,  m0b, __ATOMIC_RELAXED, __HIP_MEMORY_SCOPE_AGENT);
    float* p1 = p0 + plane;
    __hip_atomic_fetch_add(p1 + lane,       w2s * e1x, __ATOMIC_RELAXED, __HIP_MEMORY_SCOPE_AGENT);
    __hip_atomic_fetch_add(p1 + 32 + lane,  w3e * vx,  __ATOMIC_RELAXED, __HIP_MEMORY_SCOPE_AGENT);
    float* p2 = p1 + plane;
    __hip_atomic_fetch_add(p2 + lane,       w2s * e1y, __ATOMIC_RELAXED, __HIP_MEMORY_SCOPE_AGENT);
    __hip_atomic_fetch_add(p2 + 32 + lane,  w3e * vy,  __ATOMIC_RELAXED, __HIP_MEMORY_SCOPE_AGENT);
    float* p3 = p2 + plane;
    __hip_atomic_fetch_add(p3 + lane,       w2s * e1z, __ATOMIC_RELAXED, __HIP_MEMORY_SCOPE_AGENT);
    __hip_atomic_fetch_add(p3 + 32 + lane,  w3e * vz,  __ATOMIC_RELAXED, __HIP_MEMORY_SCOPE_AGENT);
}

// helper: 32B fragment load from LDS as two 16B reads (16B alignment only)
__device__ __forceinline__ v16h load_frag(const _Float16* p) {
    v8h lo = *(const v8h*)p;
    v8h hi = *(const v8h*)(p + 16);
    v16h r;
#pragma unroll
    for (int h = 0; h < 8; ++h) { r[h] = lo[h]; r[8 + h] = hi[h]; }
    return r;
}

// ---------------------------------------------------------------------------
// Kernel 3: node stage. Per wave: 16 nodes.
//   lin  = msg(16x64) @ Wl(64x32) * 1/sqrt(64)          -> 2 K-steps WMMA f16
//   out  = A2(16x320) @ WskP(320x32) * 1/sqrt(320)      -> 10 K-steps WMMA f16
// Contraction flattened kk = v*32+u (WskP[kk][w] = Wsk[u][v][w]), so K-step t
// has fixed v=t and A2 frag = lin_frag * attr[M][t] (per-lane scalar).
// ---------------------------------------------------------------------------
__global__ __launch_bounds__(256) void node_kernel(
    const float* __restrict__ node_attrs,   // [N][10]
    const float* __restrict__ msgs,         // [4][N][64]
    const float* __restrict__ Wl0,          // [64][32]
    const float* __restrict__ Wl1,          // [64][32]
    const float* __restrict__ Wsk0,         // [32][10][32]
    const float* __restrict__ Wsk1,         // [32][10][32]
    float* __restrict__ out,                // [N][128]
    int N)
{
    // transposed (+permuted, +padded) f16 weight caches
    __shared__ __align__(32) _Float16 WlS[2][32 * WL_STRIDE];   // [o][k]
    __shared__ __align__(32) _Float16 WskS[2][32 * WSK_STRIDE]; // [w][v*32+u]
    __shared__ __align__(32) _Float16 linS[8][16 * LIN_STRIDE]; // [m][u]

    int tid = threadIdx.x;
    for (int i = tid; i < 64 * 32; i += 256) {
        int k = i >> 5, o = i & 31;
        WlS[0][o * WL_STRIDE + k] = (_Float16)Wl0[i];
        WlS[1][o * WL_STRIDE + k] = (_Float16)Wl1[i];
    }
    for (int i = tid; i < 320 * 32; i += 256) {
        int u = i / 320;
        int r = i - u * 320;
        int v = r >> 5, w = r & 31;
        WskS[0][w * WSK_STRIDE + v * 32 + u] = (_Float16)Wsk0[i];
        WskS[1][w * WSK_STRIDE + v * 32 + u] = (_Float16)Wsk1[i];
    }

    int wave = tid >> 5, lane = tid & 31;
    int node_base = (blockIdx.x * 8 + wave) * 16;

    int M     = lane & 15;
    int hi    = (lane >> 4) & 1;
    int abase = hi * 8;        // A-frag K origin for this half-wave
    int bbase = hi * 8;        // B-frag K origin for this half-wave
    int nodeM = node_base + M;
    int nodeMc = (nodeM < N) ? nodeM : 0;   // clamp: EXEC stays all-1s

    // per-lane attrs row (registers; M is fixed per lane)
    _Float16 at[10];
#pragma unroll
    for (int v = 0; v < 10; ++v)
        at[v] = (_Float16)node_attrs[(size_t)nodeMc * 10 + v];

    __syncthreads();

    const float inv_sq = 0.125f;              // 1/sqrt(64)
    const float alpha  = 0.05590169943749f;   // 1/sqrt(320)

#pragma unroll
    for (int c = 0; c < 4; ++c) {
        int cc = (c == 0) ? 0 : 1;

        // ---------------- lin GEMM: (16x64)@(64x32) ----------------
        const float* msgp = msgs + (size_t)c * (size_t)N * 64;
        v8f acc0 = {}; v8f acc1 = {};
#pragma unroll
        for (int t = 0; t < 2; ++t) {
            const float4* ar = (const float4*)(msgp + (size_t)nodeMc * 64 + t * 32);
            int q = abase >> 2;                    // 0 or 2
            float4 q0 = ar[q],     q1 = ar[q + 1]; // K = abase..abase+7
            float4 q2 = ar[q + 4], q3 = ar[q + 5]; // K = 16+abase..
            v16h a;
            a[0]=(_Float16)q0.x; a[1]=(_Float16)q0.y; a[2]=(_Float16)q0.z; a[3]=(_Float16)q0.w;
            a[4]=(_Float16)q1.x; a[5]=(_Float16)q1.y; a[6]=(_Float16)q1.z; a[7]=(_Float16)q1.w;
            a[8]=(_Float16)q2.x; a[9]=(_Float16)q2.y; a[10]=(_Float16)q2.z; a[11]=(_Float16)q2.w;
            a[12]=(_Float16)q3.x; a[13]=(_Float16)q3.y; a[14]=(_Float16)q3.z; a[15]=(_Float16)q3.w;

            v16h b0 = load_frag(&WlS[cc][M        * WL_STRIDE + t * 32 + bbase]);
            v16h b1 = load_frag(&WlS[cc][(16 + M) * WL_STRIDE + t * 32 + bbase]);
            acc0 = __builtin_amdgcn_wmma_f32_16x16x32_f16(false, a, false, b0, (short)0, acc0, false, false);
            acc1 = __builtin_amdgcn_wmma_f32_16x16x32_f16(false, a, false, b1, (short)0, acc1, false, false);
        }
        // scatter lin tile (C layout: VGPR g -> M = g + hi*8, col = nt*16 + M)
#pragma unroll
        for (int g = 0; g < 8; ++g) {
            int m = g + hi * 8;
            linS[wave][m * LIN_STRIDE + M]      = (_Float16)(acc0[g] * inv_sq);
            linS[wave][m * LIN_STRIDE + 16 + M] = (_Float16)(acc1[g] * inv_sq);
        }
        __syncthreads();

        // ---- lin fragment for this lane (reused by all 10 K-steps) ----
        v8h flo = *(const v8h*)&linS[wave][M * LIN_STRIDE + abase];
        v8h fhi = *(const v8h*)&linS[wave][M * LIN_STRIDE + 16 + abase];
        v16h af;
#pragma unroll
        for (int h = 0; h < 8; ++h) { af[h] = flo[h]; af[8 + h] = fhi[h]; }

        // ---------------- skip GEMM: (16x320)@(320x32) ----------------
        v8f o0 = {}; v8f o1 = {};
#pragma unroll
        for (int t = 0; t < 10; ++t) {
            _Float16 s = at[t];
            v16h a2;
#pragma unroll
            for (int h = 0; h < 16; ++h) a2[h] = af[h] * s;
            v16h b0 = load_frag(&WskS[cc][M        * WSK_STRIDE + t * 32 + bbase]);
            v16h b1 = load_frag(&WskS[cc][(16 + M) * WSK_STRIDE + t * 32 + bbase]);
            o0 = __builtin_amdgcn_wmma_f32_16x16x32_f16(false, a2, false, b0, (short)0, o0, false, false);
            o1 = __builtin_amdgcn_wmma_f32_16x16x32_f16(false, a2, false, b1, (short)0, o1, false, false);
        }

        // store: out[:,0:32]=out0 ; out[:,32+w*3+i]=out1[:,w,i]
#pragma unroll
        for (int g = 0; g < 8; ++g) {
            int m = g + hi * 8;
            int n = node_base + m;
            if (n < N) {
                float v0 = o0[g] * alpha;
                float v1 = o1[g] * alpha;
                size_t row = (size_t)n * 128;
                if (c == 0) {
                    out[row + M]      = v0;
                    out[row + 16 + M] = v1;
                } else {
                    out[row + 32 + M * 3 + (c - 1)]        = v0;
                    out[row + 32 + (16 + M) * 3 + (c - 1)] = v1;
                }
            }
        }
        __syncthreads();  // protect linS before next channel
    }
}

// ---------------------------------------------------------------------------
extern "C" void kernel_launch(void* const* d_in, const int* in_sizes, int n_in,
                              void* d_out, int out_size, void* d_ws, size_t ws_size,
                              hipStream_t stream) {
    const float*     node_attrs = (const float*)d_in[0];
    const float*     node_feats = (const float*)d_in[1];
    const float*     edge_attrs = (const float*)d_in[2];
    const float*     edge_feats = (const float*)d_in[3];
    const long long* edge_index = (const long long*)d_in[4];
    const float*     Wr   = (const float*)d_in[5];
    const float*     Wl0  = (const float*)d_in[6];
    const float*     Wl1  = (const float*)d_in[7];
    const float*     Wsk0 = (const float*)d_in[8];
    const float*     Wsk1 = (const float*)d_in[9];
    float* out  = (float*)d_out;
    float* msgs = (float*)d_ws;

    int N = in_sizes[1] / 128;    // node_feats: [N][128]
    int E = in_sizes[3] / NRAD;   // edge_feats: [E][8]

    long n4 = ((long)4 * N * 64) / 4;
    zero_kernel<<<2048, 256, 0, stream>>>((float4*)msgs, n4);

    int eblocks = (E + 7) / 8;    // 8 edges (waves) per 256-thread block
    edge_kernel<<<eblocks, 256, 0, stream>>>(edge_attrs, edge_feats, edge_index,
                                             node_feats, Wr, msgs, E, N);

    int nblocks = (N + 127) / 128; // 128 nodes per block (8 waves x 16)
    node_kernel<<<nblocks, 256, 0, stream>>>(node_attrs, msgs, Wl0, Wl1,
                                             Wsk0, Wsk1, out, N);
}
